// DualTransformerCritic_73624329388135
// MI455X (gfx1250) — compile-verified, matrix operand order checked by reading, not run on hardware
//
#include <hip/hip_runtime.h>
#include <hip/hip_bf16.h>

typedef __bf16 bf16;
typedef __attribute__((ext_vector_type(16))) __bf16 v16bf;
typedef __attribute__((ext_vector_type(8)))  float  v8f;

#define NTHREADS 256
#define NWAVES   8

// ---------------------------------------------------------------------------
// Fragment loader for v_wmma_f32_16x16x32_bf16 (wave32).
// A-matrix 16x32 bf16 layout (ISA 7.12.2): lane = {half = lane>>4, idx = lane&15},
// dword v in 0..3 holds K = 2v,2v+1 (+8 if half); dword v in 4..7 holds
// K = 16+2(v-4), +1 (+8 if half).  B fragment is the same pattern applied to
// columns of B, so one loader serves both via (rs,ks) strides:
// element (i,k) lives at p0[i*rs + k*ks].
// ---------------------------------------------------------------------------
template <typename T>
__device__ __forceinline__ v16bf load_frag(const T* p0, int rs, int ks, int lane) {
    const int half = lane >> 4;
    const int idx  = lane & 15;
    const T* p = p0 + idx * rs + (half * 8) * ks;
    v16bf f;
#pragma unroll
    for (int v = 0; v < 8; ++v) {
        const int kb = (v < 4) ? (2 * v) : (16 + 2 * (v - 4));
        f[2 * v]     = (bf16)(float)p[kb * ks];
        f[2 * v + 1] = (bf16)(float)p[(kb + 1) * ks];
    }
    return f;
}

// ---------------------------------------------------------------------------
// Workgroup GEMM: C[M x Nt] = act( scale * (A[M x K] @ B^T) + bias ).
// A indexed (m,k) via (a_rs,a_ks); B indexed (n,k) via (b_rs,b_ks).
// Tile schedule: wave w owns column-tile n0 = (w % nt)*16 and iterates row
// tiles mi = w/nt, w/nt + NWAVES/nt, ... — so the B fragments for the whole
// K extent are loaded ONCE per wave (hoisted), and only A fragments stream
// through the m-tile loop. Requires NWAVES % (Nt/16) == 0 (true for Nt=64,128).
// ---------------------------------------------------------------------------
template <int M, int Nt, int K, bool LRELU, bool OUT_BF, typename TA, typename TB>
__device__ __forceinline__ void gemm(const TA* A, int a_rs, int a_ks,
                                     const TB* B, int b_rs, int b_ks,
                                     void* C, int ldc,
                                     const float* bias, float scale, int tid) {
    constexpr int nt  = Nt >> 4;
    constexpr int mt  = M >> 4;
    constexpr int kch = K >> 5;
    static_assert(NWAVES % nt == 0, "column tiles must divide wave count");
    constexpr int mstep = NWAVES / nt;

    const int lane = tid & 31;
    const int wave = tid >> 5;
    const int n0   = (wave % nt) << 4;

    v16bf bfrag[kch];
#pragma unroll
    for (int kc = 0; kc < kch; ++kc)
        bfrag[kc] = load_frag(B + n0 * b_rs + (kc * 32) * b_ks, b_rs, b_ks, lane);

    const int half = lane >> 4;
    const int n    = lane & 15;
    const float bv = bias ? bias[n0 + n] : 0.0f;

    for (int mi = wave / nt; mi < mt; mi += mstep) {
        const int m0 = mi << 4;
        v8f acc = {};
#pragma unroll
        for (int kc = 0; kc < kch; ++kc) {
            v16bf fa = load_frag(A + m0 * a_rs + (kc * 32) * a_ks, a_rs, a_ks, lane);
            acc = __builtin_amdgcn_wmma_f32_16x16x32_bf16(
                false, fa, false, bfrag[kc], (short)0, acc, false, false);
        }
        // C/D layout: row = m0 + v + 8*half, col = n0 + (lane&15)
#pragma unroll
        for (int v = 0; v < 8; ++v) {
            float x = acc[v] * scale + bv;
            if (LRELU) x = (x > 0.0f) ? x : 0.01f * x;
            const int row = m0 + v + 8 * half;
            if (OUT_BF) ((bf16*)C)[row * ldc + n0 + n] = (bf16)x;
            else        ((float*)C)[row * ldc + n0 + n] = x;
        }
    }
}

// Row softmax over a 64x64 fp32 LDS matrix; also mirrors result to global.
__device__ __forceinline__ void softmax64(float* S, float* out_g, int tid) {
    if (tid < 64) {
        float* r = S + tid * 64;
        float m = r[0];
        for (int j = 1; j < 64; ++j) m = fmaxf(m, r[j]);
        float s = 0.0f;
        for (int j = 0; j < 64; ++j) { float e = __expf(r[j] - m); r[j] = e; s += e; }
        const float inv = 1.0f / s;
        for (int j = 0; j < 64; ++j) {
            r[j] *= inv;
            out_g[tid * 64 + j] = r[j];
        }
    }
}

// ---------------------------------------------------------------------------
// Weight pre-conversion: pack all weight matrices as bf16 into workspace so
// GEMM B-operands are pure bf16 loads (no per-tile v_cvt in inner loops).
// Layout (element offsets): se_pre 0, key 8192, query 24576, se 40960,
// sap 57344, av 69632, f1 86016; total 94208.
// ---------------------------------------------------------------------------
__global__ __launch_bounds__(NTHREADS)
void cvt_weights_kernel(const float* __restrict__ s_se_pre, const float* __restrict__ s_key,
                        const float* __restrict__ s_query,  const float* __restrict__ s_se,
                        const float* __restrict__ s_sap,    const float* __restrict__ s_av,
                        const float* __restrict__ s_f1,     bf16* __restrict__ dst) {
    const int i = blockIdx.x * NTHREADS + threadIdx.x;
    if (i >= 94208) return;
    float v;
    if      (i < 8192)  v = s_se_pre[i];
    else if (i < 24576) v = s_key[i - 8192];
    else if (i < 40960) v = s_query[i - 24576];
    else if (i < 57344) v = s_se[i - 40960];
    else if (i < 69632) v = s_sap[i - 57344];
    else if (i < 86016) v = s_av[i - 69632];
    else                v = s_f1[i - 86016];
    dst[i] = (bf16)v;
}

// ---------------------------------------------------------------------------
// Kernel 1: one workgroup per batch. Both attention stages + S / Delta.
// LDS: 3 x (64x128 bf16) + (64x64 f32) = 64 KB.
// ---------------------------------------------------------------------------
__global__ __launch_bounds__(NTHREADS)
void critic_stage_kernel(const float* __restrict__ states,
                         const float* __restrict__ policies,
                         const float* __restrict__ actions,
                         const bf16*  __restrict__ Wbf,
                         const float* __restrict__ b_se_pre,
                         const float* __restrict__ b_se,
                         const float* __restrict__ b_sap,
                         float* __restrict__ out_wpre, float* __restrict__ out_w,
                         float* __restrict__ Sg, float* __restrict__ Dg) {
    __shared__ bf16  E [64 * 128];
    __shared__ bf16  Qb[64 * 128];
    __shared__ bf16  Kb[64 * 128];
    __shared__ float Sm[64 * 64];

    const int b   = blockIdx.x;
    const int tid = threadIdx.x;
    const float inv_sqrt_dk = 0.08838834764831845f;  // 1/sqrt(128)

    const bf16* W_se_pre = Wbf;
    const bf16* W_key    = Wbf + 8192;
    const bf16* W_query  = Wbf + 24576;
    const bf16* W_se     = Wbf + 40960;
    const bf16* W_sap    = Wbf + 57344;
    const bf16* W_av     = Wbf + 69632;

    const float* st = states   + b * (64 * 64);
    const float* po = policies + b * (64 * 32);
    const float* ac = actions  + b * (64 * 32);

    // ---- stage 1 ----
    // e = lrelu(states @ W_se_pre^T + b)            [64 x 128]
    gemm<64, 128, 64, true, true>(st, 64, 1, W_se_pre, 64, 1, E, 128, b_se_pre, 1.0f, tid);
    __syncthreads();
    // q1 = e @ Wq^T ; k1 = e @ Wk^T                 [64 x 128]
    gemm<64, 128, 128, false, true>(E, 128, 1, W_query, 128, 1, Qb, 128, nullptr, 1.0f, tid);
    gemm<64, 128, 128, false, true>(E, 128, 1, W_key,   128, 1, Kb, 128, nullptr, 1.0f, tid);
    __syncthreads();
    // scores1 = (q1 @ k1^T) / sqrt(dk)              [64 x 64] fp32
    gemm<64, 64, 128, false, false>(Qb, 128, 1, Kb, 128, 1, Sm, 64, nullptr, inv_sqrt_dk, tid);
    __syncthreads();
    softmax64(Sm, out_wpre + b * 4096, tid);   // w_pre
    __syncthreads();
    // av_pre = w_pre @ e   (B = e, column-major access) -> Qb
    gemm<64, 128, 64, false, true>(Sm, 64, 1, E, 1, 128, Qb, 128, nullptr, 1.0f, tid);
    __syncthreads();
    // se = lrelu(av_pre @ W_se^T + b_se) -> E
    gemm<64, 128, 128, true, true>(Qb, 128, 1, W_se, 128, 1, E, 128, b_se, 1.0f, tid);
    __syncthreads();
    // ---- stage 2 ----
    gemm<64, 128, 128, false, true>(E, 128, 1, W_query, 128, 1, Qb, 128, nullptr, 1.0f, tid);
    gemm<64, 128, 128, false, true>(E, 128, 1, W_key,   128, 1, Kb, 128, nullptr, 1.0f, tid);
    __syncthreads();
    gemm<64, 64, 128, false, false>(Qb, 128, 1, Kb, 128, 1, Sm, 64, nullptr, inv_sqrt_dk, tid);
    __syncthreads();
    softmax64(Sm, out_w + b * 4096, tid);      // w (kept in Sm)
    __syncthreads();

    // ---- av_act / av_pol (diagonal trick: only 2*N rows, not N*N) ----
    // oa_act = [states | actions] -> Qb (bf16, ld = 96)
    for (int idx = tid; idx < 64 * 96; idx += NTHREADS) {
        const int j = idx / 96, c = idx % 96;
        Qb[idx] = (bf16)((c < 64) ? st[j * 64 + c] : ac[j * 32 + (c - 64)]);
    }
    __syncthreads();
    gemm<64, 128, 96, true, true>(Qb, 96, 1, W_sap, 96, 1, Kb, 128, b_sap, 1.0f, tid);
    __syncthreads();
    // av_act = emb_act @ W_av^T -> E
    gemm<64, 128, 128, false, true>(Kb, 128, 1, W_av, 128, 1, E, 128, nullptr, 1.0f, tid);
    __syncthreads();
    // oa_pol -> Qb
    for (int idx = tid; idx < 64 * 96; idx += NTHREADS) {
        const int j = idx / 96, c = idx % 96;
        Qb[idx] = (bf16)((c < 64) ? st[j * 64 + c] : po[j * 32 + (c - 64)]);
    }
    __syncthreads();
    gemm<64, 128, 96, true, true>(Qb, 96, 1, W_sap, 96, 1, Kb, 128, b_sap, 1.0f, tid);
    __syncthreads();
    // av_pol = emb_pol @ W_av^T -> Qb
    gemm<64, 128, 128, false, true>(Kb, 128, 1, W_av, 128, 1, Qb, 128, nullptr, 1.0f, tid);
    __syncthreads();

    // Delta = av_pol - av_act  (fp32 to workspace)
    for (int idx = tid; idx < 64 * 128; idx += NTHREADS)
        Dg[b * 8192 + idx] = (float)Qb[idx] - (float)E[idx];
    // S = w @ av_act  (B = av_act column-major) -> workspace fp32
    gemm<64, 128, 64, false, false>(Sm, 64, 1, E, 1, 128, Sg + b * 8192, 128, nullptr, 1.0f, tid);
}

// ---------------------------------------------------------------------------
// Kernel 2: one workgroup per (b,i). node[j,:] = S[i,:] + w[i,j]*Delta[j,:],
// hidden = lrelu(node @ W_f1^T) via WMMA, value = hidden @ W_f2^T.
// ---------------------------------------------------------------------------
__global__ __launch_bounds__(NTHREADS)
void critic_head_kernel(const float* __restrict__ Sg, const float* __restrict__ Dg,
                        const float* __restrict__ w_all,
                        const bf16*  __restrict__ Wbf,
                        const float* __restrict__ W_f2,
                        float* __restrict__ out_val) {
    __shared__ bf16  Nd[64 * 128];
    __shared__ float H [64 * 64];
    __shared__ float Srow[128];
    __shared__ float wrow[64];

    const bf16* W_f1 = Wbf + 86016;

    const int blk = blockIdx.x;
    const int b   = blk >> 6;
    const int i   = blk & 63;
    const int tid = threadIdx.x;

    if (tid < 128) Srow[tid] = Sg[b * 8192 + i * 128 + tid];
    if (tid < 64)  wrow[tid] = w_all[b * 4096 + i * 64 + tid];
    __syncthreads();

    for (int idx = tid; idx < 64 * 128; idx += NTHREADS) {
        const int j = idx >> 7, d = idx & 127;
        Nd[idx] = (bf16)(Srow[d] + wrow[j] * Dg[b * 8192 + idx]);
    }
    __syncthreads();
    // hidden = lrelu(node @ W_f1^T)   [64 x 64], K = 128
    gemm<64, 64, 128, true, false>(Nd, 128, 1, W_f1, 128, 1, H, 64, nullptr, 1.0f, tid);
    __syncthreads();
    if (tid < 64) {
        float s = 0.0f;
        for (int d = 0; d < 64; ++d) s += H[tid * 64 + d] * W_f2[d];
        out_val[b * 4096 + i * 64 + tid] = s;
    }
}

// ---------------------------------------------------------------------------
extern "C" void kernel_launch(void* const* d_in, const int* in_sizes, int n_in,
                              void* d_out, int out_size, void* d_ws, size_t ws_size,
                              hipStream_t stream) {
    (void)in_sizes; (void)n_in; (void)out_size; (void)ws_size;
    const float* states   = (const float*)d_in[0];
    const float* policies = (const float*)d_in[1];
    const float* actions  = (const float*)d_in[2];
    const float* W_se_pre = (const float*)d_in[3];
    const float* b_se_pre = (const float*)d_in[4];
    const float* W_key    = (const float*)d_in[5];
    const float* W_query  = (const float*)d_in[6];
    const float* W_se     = (const float*)d_in[7];
    const float* b_se     = (const float*)d_in[8];
    const float* W_sap    = (const float*)d_in[9];
    const float* b_sap    = (const float*)d_in[10];
    const float* W_av     = (const float*)d_in[11];
    const float* W_f1     = (const float*)d_in[12];
    const float* W_f2     = (const float*)d_in[13];

    float* out      = (float*)d_out;
    float* out_val  = out;              // [B,N,N,1] = 262144
    float* out_wpre = out + 262144;     // [B,N,N]
    float* out_w    = out + 524288;     // [B,N,N]

    float* Sg  = (float*)d_ws;          // [B,64,128] fp32 = 2 MB
    float* Dg  = Sg + 64 * 64 * 128;    // [B,64,128] fp32 = 2 MB
    bf16*  Wbf = (bf16*)(Dg + 64 * 64 * 128);  // 94208 bf16 = 184 KB

    cvt_weights_kernel<<<(94208 + NTHREADS - 1) / NTHREADS, NTHREADS, 0, stream>>>(
        W_se_pre, W_key, W_query, W_se, W_sap, W_av, W_f1, Wbf);

    critic_stage_kernel<<<64, NTHREADS, 0, stream>>>(
        states, policies, actions, Wbf,
        b_se_pre, b_se, b_sap,
        out_wpre, out_w, Sg, Dg);

    critic_head_kernel<<<64 * 64, NTHREADS, 0, stream>>>(
        Sg, Dg, out_w, Wbf, W_f2, out_val);
}
